// AttentionBlock_53652731461857
// MI455X (gfx1250) — matrix operand order repaired and checked by reading, not compiled, not run on hardware
//
#include <hip/hip_runtime.h>

#define CCH 512
#define BAT 32
#define LL  4096
#define NGRP 32
#define CPG 16

typedef unsigned short u16;
typedef __attribute__((ext_vector_type(16))) __bf16 v16bf;
typedef __attribute__((ext_vector_type(8)))  float  v8f;
typedef __attribute__((ext_vector_type(4)))  unsigned int v4u;

union Frag { v16bf bf; v4u q[2]; };

__device__ __forceinline__ u16 f2bf(float f) {
  unsigned int u = __float_as_uint(f);
  u += 0x7FFFu + ((u >> 16) & 1u);   // round-to-nearest-even
  return (u16)(u >> 16);
}

// ---------------- weight f32 -> bf16 conversion ----------------
__global__ __launch_bounds__(256) void cvt_weights(const float* __restrict__ qkvw,
                                                   const float* __restrict__ projw,
                                                   u16* __restrict__ wbf,
                                                   u16* __restrict__ pwbf) {
  int i = blockIdx.x * 256 + threadIdx.x;
  const int NQ = 3 * CCH * CCH;
  if (i < NQ) {
    wbf[i] = f2bf(qkvw[i]);
  } else {
    int j = i - NQ;
    if (j < CCH * CCH) pwbf[j] = f2bf(projw[j]);
  }
}

// ---------------- GroupNorm: stats + normalize + bf16 transpose ----------------
// one block per (batch, group); writes xnT in (b, l, c) layout (c contiguous = K-dim of QKV GEMM)
__global__ __launch_bounds__(256) void groupnorm_kernel(const float* __restrict__ x,
                                                        const float* __restrict__ gw,
                                                        const float* __restrict__ gb,
                                                        float* __restrict__ stats,
                                                        u16* __restrict__ xnT) {
  __shared__ float ssum[256], ssq[256];
  __shared__ float smean, srstd;
  const int bz = blockIdx.x >> 5;
  const int g  = blockIdx.x & 31;
  const int tid = threadIdx.x;
  const size_t base = ((size_t)bz * CCH + (size_t)g * CPG) * LL;
  const int N = CPG * LL;

  float s = 0.f, s2 = 0.f;
  for (int i = tid; i < N; i += 256) { float v = x[base + i]; s += v; s2 += v * v; }
  ssum[tid] = s; ssq[tid] = s2;
  __syncthreads();
  for (int o = 128; o > 0; o >>= 1) {
    if (tid < o) { ssum[tid] += ssum[tid + o]; ssq[tid] += ssq[tid + o]; }
    __syncthreads();
  }
  if (tid == 0) {
    float mean = ssum[0] * (1.f / N);
    float var  = ssq[0] * (1.f / N) - mean * mean;
    float rstd = rsqrtf(var + 1e-5f);
    smean = mean; srstd = rstd;
    stats[(size_t)blockIdx.x * 2 + 0] = mean;
    stats[(size_t)blockIdx.x * 2 + 1] = rstd;
  }
  __syncthreads();
  const float mean = smean, rstd = srstd;
  for (int i = tid; i < N; i += 256) {
    int cl = i >> 12, l = i & (LL - 1);
    int c = g * CPG + cl;
    float xn = (x[base + i] - mean) * rstd * gw[c] + gb[c];
    xnT[((size_t)bz * LL + l) * CCH + c] = f2bf(xn);
  }
}

// ---------------- generic WMMA bf16 GEMM ----------------
// A: (M,K) bf16 row-major, ld=K.  B: (N,K) bf16 row-major (B-transposed), ld=K.
// Block = 256 threads = 8 waves; wave tile 32x64 (2x4 WMMA accumulators); block tile 128x128.
// MODE 0: QKV epilogue   MODE 1: logits f32   MODE 2: retT bf16   MODE 3: proj + residual
template <int MODE>
__global__ __launch_bounds__(256) void gemm_wmma(
    const u16* __restrict__ A, size_t sA,
    const u16* __restrict__ B, size_t sB,
    int K,
    const float* __restrict__ bias,
    u16* __restrict__ oq, u16* __restrict__ ok, u16* __restrict__ ovT,
    float* __restrict__ of32,
    u16* __restrict__ obf,
    const float* __restrict__ xsrc, const float* __restrict__ stats,
    const float* __restrict__ gw, const float* __restrict__ gb)
{
  const int tid  = threadIdx.x;
  const int lane = tid & 31;
  const int wv   = tid >> 5;
  const int half = lane >> 4;
  const int l15  = lane & 15;
  const int bz   = blockIdx.z;

  const u16* Ab = A + (size_t)bz * sA;
  const u16* Bb = B + (size_t)bz * sB;
  const int m0 = blockIdx.x * 128 + (wv >> 1) * 32;
  const int n0 = blockIdx.y * 128 + (wv & 1) * 64;

  const u16* arow[2];
  const u16* brow[4];
#pragma unroll
  for (int i = 0; i < 2; ++i) arow[i] = Ab + (size_t)(m0 + 16 * i + l15) * K;
#pragma unroll
  for (int j = 0; j < 4; ++j) brow[j] = Bb + (size_t)(n0 + 16 * j + l15) * K;

  v8f acc[2][4];
#pragma unroll
  for (int i = 0; i < 2; ++i)
#pragma unroll
    for (int j = 0; j < 4; ++j)
#pragma unroll
      for (int r = 0; r < 8; ++r) acc[i][j][r] = 0.f;

  for (int k0 = 0; k0 < K; k0 += 32) {
    Frag fa[2], fb[4];
#pragma unroll
    for (int i = 0; i < 2; ++i) {
      // A 16x32 bf16 fragment: VGPR0-3 = K in [8h, 8h+8), VGPR4-7 = K in [16+8h, 16+8h+8)
      fa[i].q[0] = *(const v4u*)(arow[i] + k0 + 8 * half);
      fa[i].q[1] = *(const v4u*)(arow[i] + k0 + 16 + 8 * half);
    }
#pragma unroll
    for (int j = 0; j < 4; ++j) {
      // B 32x16 bf16 fragment: VGPR0-3 = K in [8h, 8h+8), VGPR4-7 = K in [16+8h, 16+8h+8), fixed N
      fb[j].q[0] = *(const v4u*)(brow[j] + k0 + 8 * half);
      fb[j].q[1] = *(const v4u*)(brow[j] + k0 + 16 + 8 * half);
    }
#pragma unroll
    for (int i = 0; i < 2; ++i)
#pragma unroll
      for (int j = 0; j < 4; ++j)
        acc[i][j] = __builtin_amdgcn_wmma_f32_16x16x32_bf16(
            false, fa[i].bf, false, fb[j].bf, (short)0, acc[i][j], false, false);
  }

#pragma unroll
  for (int i = 0; i < 2; ++i) {
#pragma unroll
    for (int j = 0; j < 4; ++j) {
#pragma unroll
      for (int r = 0; r < 8; ++r) {
        const int m = m0 + 16 * i + 8 * half + r;   // D layout: VGPR r -> M = r + 8*half
        const int n = n0 + 16 * j + l15;            //           N = lane & 15
        float v = acc[i][j][r];
        if (MODE == 0) {
          v += bias[m];
          if (m < CCH) {                 // Q, fold attention scale 1/sqrt(L) = 1/64
            oq[(size_t)bz * CCH * LL + (size_t)m * LL + n] = f2bf(v * (1.0f / 64.0f));
          } else if (m < 2 * CCH) {      // K in (c,l)
            ok[(size_t)bz * CCH * LL + (size_t)(m - CCH) * LL + n] = f2bf(v);
          } else {                       // V transposed to (l,c)
            ovT[(size_t)bz * CCH * LL + (size_t)n * CCH + (m - 2 * CCH)] = f2bf(v);
          }
        } else if (MODE == 1) {
          of32[(size_t)bz * CCH * CCH + (size_t)m * CCH + n] = v;
        } else if (MODE == 2) {
          obf[(size_t)bz * CCH * LL + (size_t)n * CCH + m] = f2bf(v);  // retT (l,c)
        } else {
          const int c = m, l = n, g = c >> 4;
          const float mean = stats[((size_t)bz * NGRP + g) * 2 + 0];
          const float rstd = stats[((size_t)bz * NGRP + g) * 2 + 1];
          const size_t xi = ((size_t)bz * CCH + c) * (size_t)LL + l;
          const float xn = (xsrc[xi] - mean) * rstd * gw[c] + gb[c];
          of32[xi] = xn + v + bias[c];
        }
      }
    }
  }
}

// ---------------- softmax: one wave32 per row of 512 ----------------
__global__ __launch_bounds__(256) void softmax_kernel(const float* __restrict__ logits,
                                                      u16* __restrict__ attn) {
  const int lane = threadIdx.x & 31;
  const int row  = blockIdx.x * 8 + (threadIdx.x >> 5);
  const float* p = logits + (size_t)row * CCH;
  float v[16];
  float mx = -3.402823e38f;
#pragma unroll
  for (int t = 0; t < 16; ++t) { v[t] = p[lane + 32 * t]; mx = fmaxf(mx, v[t]); }
#pragma unroll
  for (int o = 16; o > 0; o >>= 1) mx = fmaxf(mx, __shfl_xor(mx, o, 32));
  float s = 0.f;
#pragma unroll
  for (int t = 0; t < 16; ++t) { v[t] = __expf(v[t] - mx); s += v[t]; }
#pragma unroll
  for (int o = 16; o > 0; o >>= 1) s += __shfl_xor(s, o, 32);
  const float inv = 1.f / s;
  u16* po = attn + (size_t)row * CCH;
#pragma unroll
  for (int t = 0; t < 16; ++t) po[lane + 32 * t] = f2bf(v[t] * inv);
}

// ---------------- launch ----------------
extern "C" void kernel_launch(void* const* d_in, const int* in_sizes, int n_in,
                              void* d_out, int out_size, void* d_ws, size_t ws_size,
                              hipStream_t stream) {
  (void)in_sizes; (void)n_in; (void)out_size; (void)ws_size;
  const float* x      = (const float*)d_in[0];
  const float* gn_w   = (const float*)d_in[1];
  const float* gn_b   = (const float*)d_in[2];
  const float* qkv_w  = (const float*)d_in[3];
  const float* qkv_b  = (const float*)d_in[4];
  const float* proj_w = (const float*)d_in[5];
  const float* proj_b = (const float*)d_in[6];
  float* out = (float*)d_out;

  char* ws = (char*)d_ws;
  const size_t MB = 1ull << 20;
  float* stats  = (float*)(ws + 0);            //   8 KB : (b,g) mean/rstd
  u16*   wbf    = (u16*)(ws + 1 * MB);         // 1.5 MB : qkv_w bf16
  u16*   pwbf   = (u16*)(ws + 3 * MB);         // 0.5 MB : proj_w bf16
  u16*   xnT    = (u16*)(ws + 16 * MB);        // 128 MB : xn (b,l,c) bf16
  u16*   qb     = (u16*)(ws + 144 * MB);       // 128 MB : Q (b,c,l) bf16 (pre-scaled)
  u16*   kb     = (u16*)(ws + 272 * MB);       // 128 MB : K (b,c,l) bf16
  u16*   vT     = (u16*)(ws + 400 * MB);       // 128 MB : V^T (b,l,c) bf16
  float* logits = (float*)(ws + 528 * MB);     //  32 MB : (b,q,k) f32
  u16*   attn   = (u16*)(ws + 560 * MB);       //  16 MB : (b,q,k) bf16
  u16*   retT   = (u16*)(ws + 576 * MB);       // 128 MB : ret^T (b,l,c) bf16

  const size_t sCL = (size_t)CCH * LL;
  const size_t sCC = (size_t)CCH * CCH;

  cvt_weights<<<4096, 256, 0, stream>>>(qkv_w, proj_w, wbf, pwbf);
  groupnorm_kernel<<<BAT * NGRP, 256, 0, stream>>>(x, gn_w, gn_b, stats, xnT);

  // QKV: (1536x512) x (512x4096) per batch
  gemm_wmma<0><<<dim3(12, 32, BAT), 256, 0, stream>>>(
      wbf, 0, xnT, sCL, CCH, qkv_b, qb, kb, vT,
      nullptr, nullptr, nullptr, nullptr, nullptr, nullptr);

  // logits: (512x4096) x (4096x512) per batch
  gemm_wmma<1><<<dim3(4, 4, BAT), 256, 0, stream>>>(
      qb, sCL, kb, sCL, LL, nullptr, nullptr, nullptr, nullptr,
      logits, nullptr, nullptr, nullptr, nullptr, nullptr);

  softmax_kernel<<<BAT * CCH / 8, 256, 0, stream>>>(logits, attn);

  // attn @ V: (512x512) x (512x4096) per batch
  gemm_wmma<2><<<dim3(4, 32, BAT), 256, 0, stream>>>(
      attn, sCC, vT, sCL, CCH, nullptr, nullptr, nullptr, nullptr,
      nullptr, retT, nullptr, nullptr, nullptr, nullptr);

  // proj + bias + GroupNorm residual (recomputed from x + stats)
  gemm_wmma<3><<<dim3(4, 32, BAT), 256, 0, stream>>>(
      pwbf, 0, retT, sCL, CCH, proj_b, nullptr, nullptr, nullptr,
      out, nullptr, x, stats, gn_w, gn_b);
}
